// GATDescriminator_77627238908747
// MI455X (gfx1250) — compile-verified, matrix operand order checked by reading, not run on hardware
//
#include <hip/hip_runtime.h>
#include <hip/hip_bf16.h>

#define HEADS 3
#define HID 128
#define DIN 256      // EMB + IN
#define NC1 384      // HEADS*HID (layer-1 GEMM cols)
#define LRELU_S 0.2f
#define RRELU_S 0.2291666666666667f

typedef __attribute__((ext_vector_type(16))) __bf16      v16bf;
typedef __attribute__((ext_vector_type(8)))  float       v8f;
typedef __attribute__((ext_vector_type(8)))  unsigned int v8u;

static __device__ __forceinline__ unsigned short f2bf(float f) {
    unsigned int u = __float_as_uint(f);
    u += 0x7FFFu + ((u >> 16) & 1u);          // round-to-nearest-even
    return (unsigned short)(u >> 16);
}

static __device__ __forceinline__ float lrelu(float v) {
    return v >= 0.f ? v : LRELU_S * v;
}
static __device__ __forceinline__ float rrelu(float v) {
    return v >= 0.f ? v : RRELU_S * v;
}

static __device__ __forceinline__ void atomicMaxFloat(float* addr, float val) {
    unsigned int* ua = (unsigned int*)addr;
    unsigned int old = *ua;
    while (__uint_as_float(old) < val) {
        unsigned int assumed = old;
        old = atomicCAS(ua, assumed, __float_as_uint(val));
        if (old == assumed) break;
    }
}

// ---------------- prep: build bf16 A1 = [z | x], transpose+convert weights ----
__global__ void k_prep_A1(const float* __restrict__ z, const float* __restrict__ x,
                          unsigned short* __restrict__ A1, int N) {
    int idx = blockIdx.x * blockDim.x + threadIdx.x;
    if (idx >= N * DIN) return;
    int n = idx >> 8, j = idx & 255;
    float v = (j < 128) ? z[(size_t)n * 128 + j] : x[(size_t)n * 128 + (j - 128)];
    A1[idx] = f2bf(v);
}

// W[K x Ncol] row-major -> Wt[Ncol x K] bf16
__global__ void k_transpose_w(const float* __restrict__ W, unsigned short* __restrict__ Wt,
                              int K, int Ncol) {
    int idx = blockIdx.x * blockDim.x + threadIdx.x;
    if (idx >= K * Ncol) return;
    int n = idx / K, k = idx % K;
    Wt[idx] = f2bf(W[(size_t)k * Ncol + n]);
}

// ---------------- WMMA bf16 GEMM: C[M x Ncol] = A[M x K] * Bt[Ncol x K]^T ----
// block = 128 threads (4 waves); wave w handles col tile (blockIdx.y*4 + w)
__global__ void k_gemm_bf16(const unsigned short* __restrict__ A,
                            const unsigned short* __restrict__ Bt,
                            float* __restrict__ C, int M, int K, int Ncol) {
    int lane = threadIdx.x & 31;
    int wave = threadIdx.x >> 5;
    int row0 = blockIdx.x * 16;
    int col0 = (blockIdx.y * 4 + wave) * 16;
    if (row0 >= M || col0 >= Ncol) return;   // wave-uniform
    int half = lane >> 4;
    int r    = lane & 15;

    const unsigned short* arow = A  + (size_t)(row0 + r) * K;
    const unsigned short* brow = Bt + (size_t)(col0 + r) * K;

    v8f acc = {};
    for (int k0 = 0; k0 < K; k0 += 32) {
        v8u au, bu;
#pragma unroll
        for (int v = 0; v < 8; ++v) {
            int kk = k0 + ((v >= 4) ? 16 : 0) + half * 8 + (v & 3) * 2;
            au[v] = *(const unsigned int*)(arow + kk);
            bu[v] = *(const unsigned int*)(brow + kk);
        }
        v16bf a = __builtin_bit_cast(v16bf, au);
        v16bf b = __builtin_bit_cast(v16bf, bu);
        acc = __builtin_amdgcn_wmma_f32_16x16x32_bf16(false, a, false, b,
                                                      (short)0, acc, false, false);
    }
#pragma unroll
    for (int v = 0; v < 8; ++v) {
        int mrow = row0 + v + half * 8;
        C[(size_t)mrow * Ncol + col0 + r] = acc[v];
    }
}

// ---------------- attention passes (shared by both layers) -------------------
// per (node, head): attention logits; m initialized with self-loop logit
__global__ void k_node_pre(const float* __restrict__ Hraw,
                           const float* __restrict__ a_src, const float* __restrict__ a_dst,
                           float* __restrict__ esrc, float* __restrict__ edst,
                           float* __restrict__ mbuf, int N) {
    int idx = blockIdx.x * blockDim.x + threadIdx.x;   // n*3 + h
    if (idx >= N * HEADS) return;
    int n = idx / HEADS, h = idx % HEADS;
    const float* hp = Hraw + (size_t)n * NC1 + h * HID;
    const float* as = a_src + h * HID;
    const float* ad = a_dst + h * HID;
    float ss = 0.f, sd = 0.f;
    for (int c = 0; c < HID; ++c) { float v = hp[c]; ss += v * as[c]; sd += v * ad[c]; }
    esrc[idx] = ss;
    edst[idx] = sd;
    mbuf[idx] = lrelu(ss + sd);                        // self-loop logit seeds the max
}

__global__ void k_edge_max(const int* __restrict__ src, const int* __restrict__ dst,
                           const float* __restrict__ esrc, const float* __restrict__ edst,
                           float* __restrict__ mbuf, int E) {
    int e = blockIdx.x * blockDim.x + threadIdx.x;
    if (e >= E) return;
    int s = src[e], d = dst[e];
#pragma unroll
    for (int h = 0; h < HEADS; ++h) {
        float v = lrelu(esrc[s * HEADS + h] + edst[d * HEADS + h]);
        atomicMaxFloat(&mbuf[d * HEADS + h], v);
    }
}

// self-loop contribution: denom = exp(e_self - m), acc = that * h[n]
__global__ void k_node_init(const float* __restrict__ Hraw,
                            const float* __restrict__ esrc, const float* __restrict__ edst,
                            const float* __restrict__ mbuf,
                            float* __restrict__ denom, float* __restrict__ acc, int N) {
    int idx = blockIdx.x * blockDim.x + threadIdx.x;   // n*384 + j
    if (idx >= N * NC1) return;
    int n = idx / NC1, j = idx % NC1;
    int h = j / HID, c = j % HID;
    float es = lrelu(esrc[n * HEADS + h] + edst[n * HEADS + h]);
    float w  = __expf(es - mbuf[n * HEADS + h]);
    acc[idx] = w * Hraw[idx];
    if (c == 0) denom[n * HEADS + h] = w;
}

// per (edge, channel): accumulate exp-weighted source features into dst
__global__ void k_edge_agg(const int* __restrict__ src, const int* __restrict__ dst,
                           const float* __restrict__ Hraw,
                           const float* __restrict__ esrc, const float* __restrict__ edst,
                           const float* __restrict__ mbuf,
                           float* __restrict__ denom, float* __restrict__ acc,
                           long long E) {
    long long t = (long long)blockIdx.x * blockDim.x + threadIdx.x;   // e*128 + c
    if (t >= E * (long long)HID) return;
    int e = (int)(t >> 7);
    int c = (int)(t & 127);
    int s = src[e], d = dst[e];
#pragma unroll
    for (int h = 0; h < HEADS; ++h) {
        float v  = lrelu(esrc[s * HEADS + h] + edst[d * HEADS + h]);
        float ex = __expf(v - mbuf[d * HEADS + h]);
        if (c == 0) atomicAdd(&denom[d * HEADS + h], ex);
        atomicAdd(&acc[(size_t)d * NC1 + h * HID + c],
                  ex * Hraw[(size_t)s * NC1 + h * HID + c]);
    }
}

// layer 1 finalize: mean over heads, +b1, rrelu, convert to bf16 A2
__global__ void k_finalize1(const float* __restrict__ acc, const float* __restrict__ denom,
                            const float* __restrict__ b1,
                            unsigned short* __restrict__ A2, int N) {
    int idx = blockIdx.x * blockDim.x + threadIdx.x;   // n*128 + c
    if (idx >= N * HID) return;
    int n = idx / HID, c = idx % HID;
    float s = 0.f;
#pragma unroll
    for (int h = 0; h < HEADS; ++h)
        s += acc[(size_t)n * NC1 + h * HID + c] / denom[n * HEADS + h];
    s = s * (1.0f / HEADS) + b1[c];
    A2[idx] = f2bf(rrelu(s));
}

// layer 2 finalize (concat=True): normalize, +b2, rrelu, in place into acc
__global__ void k_finalize2(float* __restrict__ acc, const float* __restrict__ denom,
                            const float* __restrict__ b2, int N) {
    int idx = blockIdx.x * blockDim.x + threadIdx.x;   // n*384 + j
    if (idx >= N * NC1) return;
    int n = idx / NC1, j = idx % NC1;
    int h = j / HID;
    float v = acc[idx] / denom[n * HEADS + h] + b2[j];
    acc[idx] = rrelu(v);
}

// final linear + sigmoid
__global__ void k_final_lin(const float* __restrict__ hfin, const float* __restrict__ Wlin,
                            const float* __restrict__ blin, float* __restrict__ out, int N) {
    int n = blockIdx.x * blockDim.x + threadIdx.x;
    if (n >= N) return;
    float s = blin[0];
    const float* hp = hfin + (size_t)n * NC1;
    for (int j = 0; j < NC1; ++j) s += hp[j] * Wlin[j];
    out[n] = 1.0f / (1.0f + __expf(-s));
}

#define CDIV(a, b) (((a) + (b) - 1) / (b))
static inline size_t alignup(size_t x) { return (x + 255) & ~(size_t)255; }

extern "C" void kernel_launch(void* const* d_in, const int* in_sizes, int n_in,
                              void* d_out, int out_size, void* d_ws, size_t ws_size,
                              hipStream_t stream) {
    const float* z      = (const float*)d_in[0];
    const float* x      = (const float*)d_in[1];
    const int*   ei     = (const int*)  d_in[2];
    const float* W1     = (const float*)d_in[3];
    const float* a1_src = (const float*)d_in[4];
    const float* a1_dst = (const float*)d_in[5];
    const float* b1     = (const float*)d_in[6];
    const float* W2     = (const float*)d_in[7];
    const float* a2_src = (const float*)d_in[8];
    const float* a2_dst = (const float*)d_in[9];
    const float* b2     = (const float*)d_in[10];
    const float* Wlin   = (const float*)d_in[11];
    const float* blin   = (const float*)d_in[12];

    const int N = in_sizes[0] / 128;        // 50000 (divisible by 16)
    const int E = in_sizes[2] / 2;          // 400000
    const int* src = ei;
    const int* dst = ei + E;

    // ---- workspace layout ----
    char* ws = (char*)d_ws;
    size_t off = 0;
    unsigned short* A1   = (unsigned short*)(ws + off); off = alignup(off + (size_t)N * DIN * 2);
    unsigned short* W1t  = (unsigned short*)(ws + off); off = alignup(off + (size_t)NC1 * DIN * 2);
    unsigned short* W2t  = (unsigned short*)(ws + off); off = alignup(off + (size_t)NC1 * HID * 2);
    float* Hraw          = (float*)(ws + off);          off = alignup(off + (size_t)N * NC1 * 4);
    float* acc           = (float*)(ws + off);          off = alignup(off + (size_t)N * NC1 * 4);
    float* esrc          = (float*)(ws + off);          off = alignup(off + (size_t)N * HEADS * 4);
    float* edst          = (float*)(ws + off);          off = alignup(off + (size_t)N * HEADS * 4);
    float* mbuf          = (float*)(ws + off);          off = alignup(off + (size_t)N * HEADS * 4);
    float* denom         = (float*)(ws + off);          off = alignup(off + (size_t)N * HEADS * 4);
    unsigned short* A2   = (unsigned short*)(ws + off); off = alignup(off + (size_t)N * HID * 2);
    (void)ws_size;

    const int TB = 256;

    // ---- prep ----
    k_prep_A1<<<CDIV((size_t)N * DIN, TB), TB, 0, stream>>>(z, x, A1, N);
    k_transpose_w<<<CDIV(DIN * NC1, TB), TB, 0, stream>>>(W1, W1t, DIN, NC1);
    k_transpose_w<<<CDIV(HID * NC1, TB), TB, 0, stream>>>(W2, W2t, HID, NC1);

    // ---- layer 1: GEMM (WMMA bf16) + GAT attention ----
    {
        dim3 grid(N / 16, NC1 / 64);
        k_gemm_bf16<<<grid, 128, 0, stream>>>(A1, W1t, Hraw, N, DIN, NC1);
    }
    k_node_pre<<<CDIV(N * HEADS, TB), TB, 0, stream>>>(Hraw, a1_src, a1_dst, esrc, edst, mbuf, N);
    k_edge_max<<<CDIV(E, TB), TB, 0, stream>>>(src, dst, esrc, edst, mbuf, E);
    k_node_init<<<CDIV((size_t)N * NC1, TB), TB, 0, stream>>>(Hraw, esrc, edst, mbuf, denom, acc, N);
    k_edge_agg<<<CDIV((long long)E * HID, TB), TB, 0, stream>>>(src, dst, Hraw, esrc, edst, mbuf,
                                                                denom, acc, (long long)E);
    k_finalize1<<<CDIV(N * HID, TB), TB, 0, stream>>>(acc, denom, b1, A2, N);

    // ---- layer 2: GEMM (WMMA bf16) + GAT attention (concat) ----
    {
        dim3 grid(N / 16, NC1 / 64);
        k_gemm_bf16<<<grid, 128, 0, stream>>>(A2, W2t, Hraw, N, HID, NC1);
    }
    k_node_pre<<<CDIV(N * HEADS, TB), TB, 0, stream>>>(Hraw, a2_src, a2_dst, esrc, edst, mbuf, N);
    k_edge_max<<<CDIV(E, TB), TB, 0, stream>>>(src, dst, esrc, edst, mbuf, E);
    k_node_init<<<CDIV((size_t)N * NC1, TB), TB, 0, stream>>>(Hraw, esrc, edst, mbuf, denom, acc, N);
    k_edge_agg<<<CDIV((long long)E * HID, TB), TB, 0, stream>>>(src, dst, Hraw, esrc, edst, mbuf,
                                                                denom, acc, (long long)E);
    k_finalize2<<<CDIV((size_t)N * NC1, TB), TB, 0, stream>>>(acc, denom, b2, N);

    // ---- head ----
    k_final_lin<<<CDIV(N, TB), TB, 0, stream>>>(acc, Wlin, blin, (float*)d_out, N);
}
